// LSTMinc_v2_77781857730586
// MI455X (gfx1250) — compile-verified, matrix operand order checked by reading, not run on hardware
//
#include <hip/hip_runtime.h>
#include <hip/hip_bf16.h>

// CDNA5 / gfx1250: wave32, WMMA 16x16x32 f16 -> f32.
typedef __attribute__((ext_vector_type(16))) _Float16 v16h;
typedef __attribute__((ext_vector_type(8)))  float    v8f;

union Frag { v16h h; unsigned int u[8]; };

__device__ __forceinline__ unsigned int pack2(float a, float b) {
    union { _Float16 h[2]; unsigned int u; } q;
    q.h[0] = (_Float16)a; q.h[1] = (_Float16)b;
    return q.u;
}

// Fast activations on the hardware TRANS pipe (v_exp_f32 / v_rcp_f32),
// which co-executes with the WMMA/XDL stream. No IEEE-divide expansion.
__device__ __forceinline__ float tanh_fast(float x) {
    x = fminf(fmaxf(x, -15.f), 15.f);                       // v_med3
    float e = __builtin_amdgcn_exp2f(x * 2.885390081777927f); // e^(2x)
    float r = __builtin_amdgcn_rcpf(e + 1.f);
    return fmaf(-2.f, r, 1.f);                              // 1 - 2/(e^(2x)+1)
}
__device__ __forceinline__ float sigf(float x) {
    x = fminf(fmaxf(x, -30.f), 30.f);
    float e = __builtin_amdgcn_exp2f(x * -1.4426950408889634f); // e^-x
    return __builtin_amdgcn_rcpf(e + 1.f);
}

__device__ __forceinline__ v8f wmma_acc(v16h a, v16h b, v8f c) {
    // D = A(16x32 f16) * B(32x16 f16) + C(16x16 f32); C==0 folds to inline 0.
    return __builtin_amdgcn_wmma_f32_16x16x32_f16(
        /*neg_a=*/false, a, /*neg_b=*/false, b,
        /*c_mod=*/(short)0, c, /*reuse_a=*/false, /*reuse_b=*/false);
}

// Load a pre-swizzled B fragment: 8 contiguous dwords per lane.
__device__ __forceinline__ v16h ldfrag(const unsigned int* p) {
    Frag f;
#pragma unroll
    for (int v = 0; v < 8; ++v) f.u[v] = p[v];
    return f.h;
}

// Build an A fragment (16x32 f16) from per-row packed f16-pair arrays in LDS.
// A layout: lanes 0-15 hold M=lane, K-pairs {0..3, 8..11}; lanes 16-31 hold
// M=lane-16, K-pairs {4..7, 12..15}.
__device__ __forceinline__ v16h ldafrag(const unsigned int* rowPairs, int hi4, int off) {
    Frag f;
#pragma unroll
    for (int v = 0; v < 8; ++v)
        f.u[v] = rowPairs[off + (((v < 4) ? v : v + 4) + hi4)];
    return f.h;
}

// One LSTM step for a 16-sample tile. Gates (16x256) computed as 16 WMMA
// tiles (bias folded into W_ih^T row k=12, x[12]==1); processed in 4
// column-groups so only 4 accumulators are live. Writes new h (f16) into sH.
__device__ __forceinline__ void lstm_step(
    v16h xf, v16h hf0, v16h hf1, float c[4][8], int lane,
    const unsigned int* sWih, const unsigned int* sWhh,
    _Float16 (*sH)[64]) {
    const int col   = lane & 15;
    const int rbase = (lane >> 4) << 3; // C/D rows: v (+8 for upper half-wave)
    const v8f z8 = {};                  // -> inline 0 SRC2
#pragma unroll
    for (int jt = 0; jt < 4; ++jt) {
        v8f acc[4];
#pragma unroll
        for (int g = 0; g < 4; ++g) {
            const int tile = g * 4 + jt;           // i:0-3 f:4-7 g:8-11 o:12-15
            v8f a = wmma_acc(xf,  ldfrag(sWih + ((size_t)tile * 32 + lane) * 8), z8);
            a = wmma_acc(hf0, ldfrag(sWhh + (((size_t)tile * 2 + 0) * 32 + lane) * 8), a);
            a = wmma_acc(hf1, ldfrag(sWhh + (((size_t)tile * 2 + 1) * 32 + lane) * 8), a);
            acc[g] = a;
        }
#pragma unroll
        for (int v = 0; v < 8; ++v) {
            float gi = acc[0][v], gf = acc[1][v], gg = acc[2][v], go = acc[3][v];
            float cv = fmaf(sigf(gf), c[jt][v], sigf(gi) * tanh_fast(gg));
            c[jt][v] = cv;
            float hv = sigf(go) * tanh_fast(cv);
            sH[rbase + v][jt * 16 + col] = (_Float16)hv;
        }
    }
}

#define WAIT_DS() asm volatile("s_wait_dscnt 0" ::: "memory")

__global__ __launch_bounds__(32) void lstm_gen_kernel(
    const float* __restrict__ noise, const float* __restrict__ hist_x,
    const float* __restrict__ gap,   const float* __restrict__ W_ih,
    const float* __restrict__ W_hh,  const float* __restrict__ b_ih,
    const float* __restrict__ b_hh,  const float* __restrict__ W1,
    const float* __restrict__ b1,    const float* __restrict__ W2,
    const float* __restrict__ b2,    float* __restrict__ out) {

    // Pre-swizzled B fragments (8 contiguous dwords per lane per tile/kchunk)
    __shared__ unsigned int sWih[16 * 32 * 8];       // W_ih^T K pad 12->32, bias at k=12
    __shared__ unsigned int sWhh[16 * 2 * 32 * 8];   // W_hh^T, K=64 (2 chunks)
    __shared__ unsigned int sW1 [4 * 2 * 32 * 8];    // W1^T,   K=64, N=64
    __shared__ unsigned int sW2B[2 * 32 * 8];        // W2^T as 64x16 B, col 0 only
    __shared__ float sB1[64];
    __shared__ float sDP[16];
    __shared__ __align__(16) unsigned int sX[16][16];  // input feats, f16 pairs (K=32)
    __shared__ __align__(16) _Float16     sH[16][64];  // hidden state h
    __shared__ __align__(16) _Float16     sM[16][64];  // MLP hidden

    const int tid = threadIdx.x;

    // ---- one-time weight swizzle into LDS (B-fragment layout) ----
    for (int idx = tid; idx < 16 * 32 * 8; idx += 32) {
        int tile = idx >> 8, lane = (idx >> 3) & 31, v = idx & 7;
        int n = tile * 16 + (lane & 15);
        int p = (((v < 4) ? v : v + 4) + ((lane >> 4) << 2));
        int k = 2 * p;                               // even
        float lo = (k < 12) ? W_ih[n * 12 + k]
                 : (k == 12 ? (b_ih[n] + b_hh[n]) : 0.f);  // fold bias at k=12
        float hi = (k + 1 < 12) ? W_ih[n * 12 + k + 1] : 0.f;
        sWih[idx] = pack2(lo, hi);
    }
    for (int idx = tid; idx < 16 * 2 * 32 * 8; idx += 32) {
        int tile = idx >> 9, kc = (idx >> 8) & 1, lane = (idx >> 3) & 31, v = idx & 7;
        int n = tile * 16 + (lane & 15);
        int p = (((v < 4) ? v : v + 4) + ((lane >> 4) << 2));
        int k = kc * 32 + 2 * p;
        sWhh[idx] = pack2(W_hh[n * 64 + k], W_hh[n * 64 + k + 1]);
    }
    for (int idx = tid; idx < 4 * 2 * 32 * 8; idx += 32) {
        int tile = idx >> 9, kc = (idx >> 8) & 1, lane = (idx >> 3) & 31, v = idx & 7;
        int n = tile * 16 + (lane & 15);
        int p = (((v < 4) ? v : v + 4) + ((lane >> 4) << 2));
        int k = kc * 32 + 2 * p;
        sW1[idx] = pack2(W1[n * 64 + k], W1[n * 64 + k + 1]);
    }
    for (int idx = tid; idx < 2 * 32 * 8; idx += 32) { // W2: only column n==0 nonzero
        int kc = idx >> 8, lane = (idx >> 3) & 31, v = idx & 7;
        int n = lane & 15;
        int p = (((v < 4) ? v : v + 4) + ((lane >> 4) << 2));
        int k = kc * 32 + 2 * p;
        sW2B[idx] = (n == 0) ? pack2(W2[k], W2[k + 1]) : 0u;
    }
    for (int i = tid; i < 64;  i += 32) sB1[i] = b1[i];
    for (int i = tid; i < 256; i += 32) ((unsigned int*)sX)[i] = 0u; // K-pad stays 0
    for (int i = tid; i < 512; i += 32) ((unsigned int*)sH)[i] = 0u; // h0 = 0
    if (tid < 16) sX[tid][6] = pack2(1.f, 0.f);   // constant feature 12 = 1 (bias)
    __syncthreads();

    const int lane  = tid;
    const int m     = lane & 15;            // sample within tile / A-row
    const int hi4   = (lane >> 4) << 2;     // A-layout pair-group offset
    const int col   = m;                    // C/D column
    const int rbase = (lane >> 4) << 3;     // C/D row base
    const bool owner = lane < 16;
    const size_t bS = (size_t)blockIdx.x * 16 + m;  // this lane's sample (owner)

    float c[4][8] = {};
    float dist = 0.f;
    const float b2v = b2[0];
    const v8f z8 = {};

    // ================= conditioning phase: t = 0..63 =================
    for (int t = 0; t < 64; ++t) {
        if (owner) {
            const float* hx = hist_x + (bS * 64 + t) * 3;
            float h0v = hx[0], h1v = hx[1], h2v = hx[2];
            dist += h2v;                                  // cumsum of last feature
            const float* nz = noise + (bS * 256 + t) * 8;
            if (t + 1 < 64) __builtin_prefetch(noise + (bS * 256 + t + 1) * 8, 0, 0);
            sX[m][0] = pack2(h0v, h1v);
            sX[m][1] = pack2(h2v, dist);
            sX[m][2] = pack2(nz[0], nz[1]);
            sX[m][3] = pack2(nz[2], nz[3]);
            sX[m][4] = pack2(nz[4], nz[5]);
            sX[m][5] = pack2(nz[6], nz[7]);
            float* op = out + (bS * 257 + t) * 3;         // copy hist to output
            op[0] = h0v; op[1] = h1v; op[2] = h2v;
        }
        WAIT_DS();   // covers sX stores (this iter) + sH stores (prev iter)
        v16h xf  = ldafrag(&sX[m][0], hi4, 0);
        const unsigned int* hp = (const unsigned int*)&sH[m][0];
        v16h hf0 = ldafrag(hp, hi4, 0);
        v16h hf1 = ldafrag(hp, hi4, 16);
        lstm_step(xf, hf0, hf1, c, lane, sWih, sWhh, sH);
    }

    // ================= generation phase: tg = 0..192 =================
    for (int tg = 0; tg < 193; ++tg) {
        WAIT_DS();   // previous step's sH stores
        // old h as A fragments (shared by MLP head and LSTM update)
        const unsigned int* hp = (const unsigned int*)&sH[m][0];
        v16h hf0 = ldafrag(hp, hi4, 0);
        v16h hf1 = ldafrag(hp, hi4, 16);

        // MLP layer 1: tanh(h @ W1^T + b1) -> sM (16x64)
#pragma unroll
        for (int nt = 0; nt < 4; ++nt) {
            v8f a = wmma_acc(hf0, ldfrag(sW1 + (((size_t)nt * 2 + 0) * 32 + lane) * 8), z8);
            a = wmma_acc(hf1, ldfrag(sW1 + (((size_t)nt * 2 + 1) * 32 + lane) * 8), a);
            const float bv = sB1[nt * 16 + col];
#pragma unroll
            for (int v = 0; v < 8; ++v)
                sM[rbase + v][nt * 16 + col] = (_Float16)tanh_fast(a[v] + bv);
        }
        WAIT_DS();

        // MLP layer 2 via WMMA: (16x64) @ (64x16, col0=W2) -> column 0
        {
            const unsigned int* mp = (const unsigned int*)&sM[m][0];
            v16h mf0 = ldafrag(mp, hi4, 0);
            v16h mf1 = ldafrag(mp, hi4, 16);
            v8f a2 = wmma_acc(mf0, ldfrag(sW2B + ((size_t)0 * 32 + lane) * 8), z8);
            a2 = wmma_acc(mf1, ldfrag(sW2B + ((size_t)1 * 32 + lane) * 8), a2);
            if (col == 0) {                       // lanes 0 and 16 hold column 0
#pragma unroll
                for (int v = 0; v < 8; ++v)
                    sDP[rbase + v] = tanh_fast(a2[v] + b2v) * 24.0f;  // dp
            }
        }
        WAIT_DS();

        if (owner) {
            float dp = sDP[m];
            const float* gp = gap + (bS * 193 + tg) * 2;
            float g0 = gp[0], g1 = gp[1];
            dist += dp;
            float n0=0.f,n1=0.f,n2=0.f,n3=0.f,n4=0.f,n5=0.f,n6=0.f,n7=0.f;
            if (tg < 192) {
                const float* nz = noise + (bS * 256 + 64 + tg) * 8;
                n0=nz[0]; n1=nz[1]; n2=nz[2]; n3=nz[3];
                n4=nz[4]; n5=nz[5]; n6=nz[6]; n7=nz[7];
                if (tg + 1 < 192)
                    __builtin_prefetch(noise + (bS * 256 + 65 + tg) * 8, 0, 0);
            }
            sX[m][0] = pack2(g0, g1);
            sX[m][1] = pack2(dp, dist);
            sX[m][2] = pack2(n0, n1);
            sX[m][3] = pack2(n2, n3);
            sX[m][4] = pack2(n4, n5);
            sX[m][5] = pack2(n6, n7);
            float* op = out + (bS * 257 + 64 + tg) * 3;   // generated x
            op[0] = g0; op[1] = g1; op[2] = dp;
        }
        WAIT_DS();
        v16h xf = ldafrag(&sX[m][0], hi4, 0);
        lstm_step(xf, hf0, hf1, c, lane, sWih, sWhh, sH);
    }
}

extern "C" void kernel_launch(void* const* d_in, const int* in_sizes, int n_in,
                              void* d_out, int out_size, void* d_ws, size_t ws_size,
                              hipStream_t stream) {
    const float* noise  = (const float*)d_in[0];
    const float* hist_x = (const float*)d_in[1];
    const float* gap    = (const float*)d_in[2];
    const float* W_ih   = (const float*)d_in[3];
    const float* W_hh   = (const float*)d_in[4];
    const float* b_ih   = (const float*)d_in[5];
    const float* b_hh   = (const float*)d_in[6];
    const float* W1     = (const float*)d_in[7];
    const float* b1     = (const float*)d_in[8];
    const float* W2     = (const float*)d_in[9];
    const float* b2     = (const float*)d_in[10];

    const int B = in_sizes[1] / (64 * 3);   // hist_x is (B, 64, 3)
    dim3 grid((B + 15) / 16);               // one wave32 per 16-sample tile
    lstm_gen_kernel<<<grid, 32, 0, stream>>>(
        noise, hist_x, gap, W_ih, W_hh, b_ih, b_hh, W1, b1, W2, b2,
        (float*)d_out);
}
